// mhpa_78280073937596
// MI455X (gfx1250) — compile-verified, matrix operand order checked by reading, not run on hardware
//
#include <hip/hip_runtime.h>
#include <hip/hip_bf16.h>
#include <math.h>

// Problem constants (reference: B=2, N=2048, H=8, HID=256, V=32)
#define BB 2
#define NN 2048
#define NH 8
#define HID 256
#define VD 32

typedef __attribute__((ext_vector_type(16))) __bf16        v16bf;
typedef __attribute__((ext_vector_type(8)))  float         v8f;
typedef __attribute__((ext_vector_type(8)))  unsigned int  v8u;

// LDS row stride (ushorts): 80 bytes -> rows stay 16B-aligned, banks perturbed
#define LROW 40

// ---------- bf16 helpers ------------------------------------------------------
__device__ __forceinline__ unsigned short f2bfu(float f) {
    unsigned int u = __float_as_uint(f);
    u += 0x7fffu + ((u >> 16) & 1u);          // round-to-nearest-even
    return (unsigned short)(u >> 16);
}

// assemble a 16-element bf16 fragment from two 128-bit loads
__device__ __forceinline__ v16bf mk16(uint4 lo, uint4 hi) {
    v8u t = { lo.x, lo.y, lo.z, lo.w, hi.x, hi.y, hi.z, hi.w };
    return __builtin_bit_cast(v16bf, t);
}

// ---------- CDNA5 WMMA wrapper (codegen-confirmed builtin) --------------------
__device__ __forceinline__ v8f wmma_bf16(v16bf a, v16bf b, v8f c) {
    return __builtin_amdgcn_wmma_f32_16x16x32_bf16(false, a, false, b, (short)0, c,
                                                   false, false);
}

// wave-internal LDS fence: CDNA5 split dependency counters
__device__ __forceinline__ void lds_wave_fence() {
    __builtin_amdgcn_wave_barrier();
    asm volatile("s_wait_dscnt 0" ::: "memory");
    __builtin_amdgcn_wave_barrier();
}

__device__ __forceinline__ float gelu_exact(float x) {
    return 0.5f * x * (1.0f + erff(x * 0.70710678118654752440f));
}

// =============================================================================
// Kernel 1: per-row exact median (quantile 0.5, linear interp) + row min.
// d >= 0 so float bits are order-isomorphic to uint bits -> byte radix select.
// =============================================================================
__global__ __launch_bounds__(256) void row_median_min_kernel(
    const float* __restrict__ md, float* __restrict__ thr, float* __restrict__ dmin) {
    __shared__ unsigned int keys[NN];
    __shared__ unsigned int hist[256];
    __shared__ unsigned int s_minbits;
    __shared__ unsigned int s_selbin;
    __shared__ unsigned int s_k;

    const int row = blockIdx.x;
    const int tid = threadIdx.x;
    const float* rp = md + (size_t)row * NN;

    if (tid == 0) s_minbits = 0xFFFFFFFFu;
    __syncthreads();

    unsigned int lmin = 0xFFFFFFFFu;
    for (int t = tid; t < NN; t += 256) {
        unsigned int k = __float_as_uint(rp[t]);
        keys[t] = k;
        lmin = min(lmin, k);
    }
    atomicMin(&s_minbits, lmin);
    __syncthreads();

    float vals[2];
    for (int sel = 0; sel < 2; ++sel) {
        unsigned int prefix = 0, pmask = 0;
        if (tid == 0) s_k = 1023u + (unsigned)sel;   // k-th smallest, 0-indexed
        for (int shift = 24; shift >= 0; shift -= 8) {
            hist[tid] = 0;
            __syncthreads();
            for (int t = tid; t < NN; t += 256) {
                unsigned int k = keys[t];
                if ((k & pmask) == prefix)
                    atomicAdd(&hist[(k >> shift) & 255u], 1u);
            }
            __syncthreads();
            if (tid == 0) {
                unsigned int kk = s_k, run = 0, bin = 0;
                for (; bin < 256u; ++bin) {
                    unsigned int c = hist[bin];
                    if (run + c > kk) break;
                    run += c;
                }
                s_selbin = bin;
                s_k = kk - run;
            }
            __syncthreads();
            prefix |= (s_selbin << shift);
            pmask  |= (0xFFu << shift);
            __syncthreads();
        }
        vals[sel] = __uint_as_float(prefix);
    }
    if (tid == 0) {
        thr[row]  = 0.5f * (vals[0] + vals[1]);      // torch.quantile linear interp
        dmin[row] = __uint_as_float(s_minbits);
    }
}

// =============================================================================
// Kernel 2: value (TRANSPOSED out): val[b,h,v,n] = sum_j inputs[b,n,j]*weight[h,j,v]
// Block = (b, 16-row tile); 8 waves = 8 heads; all fragments are b128 LDS loads.
// =============================================================================
__global__ __launch_bounds__(256) void value_gemm_kernel(
    const float* __restrict__ inp, const float* __restrict__ wgt,
    unsigned short* __restrict__ val) {
    __shared__ unsigned short a_sh[16][LROW];         // bf16 inputs tile [m][k]
    __shared__ unsigned short w_shT[NH][32][LROW];    // bf16 weight tile  [v][k]

    const int b    = blockIdx.x / (NN / 16);
    const int mt   = blockIdx.x % (NN / 16);
    const int row0 = mt * 16;
    const int wave = threadIdx.x >> 5;               // == head h
    const int lane = threadIdx.x & 31;
    const int lm   = lane & 15;
    const int hi   = lane >> 4;
    const int base = hi << 3;                        // K offset owned by this half

    v8f acc0 = {}, acc1 = {};

    for (int kk = 0; kk < HID; kk += 32) {
        for (int t = threadIdx.x; t < 512; t += 256) {
            int m = t >> 5, k = t & 31;
            a_sh[m][k] = f2bfu(inp[((size_t)b * NN + row0 + m) * HID + kk + k]);
        }
        for (int t = 0; t < 32; ++t)                 // transpose on the fly: [v][k]
            w_shT[wave][lane][t] =
                f2bfu(wgt[((size_t)wave * HID + kk + t) * VD + lane]);
        __syncthreads();

        v16bf af = mk16(*(const uint4*)&a_sh[lm][base],
                        *(const uint4*)&a_sh[lm][16 + base]);
        v16bf b0 = mk16(*(const uint4*)&w_shT[wave][lm][base],
                        *(const uint4*)&w_shT[wave][lm][16 + base]);
        v16bf b1 = mk16(*(const uint4*)&w_shT[wave][16 + lm][base],
                        *(const uint4*)&w_shT[wave][16 + lm][16 + base]);
        acc0 = wmma_bf16(af, b0, acc0);
        acc1 = wmma_bf16(af, b1, acc1);
        __syncthreads();
    }

    // D layout: vgpr g, lane L -> m = g + 8*hi, n = lm ; store transposed [v][n]
#pragma unroll
    for (int g = 0; g < 8; ++g) {
        int m = g + (hi << 3);
        size_t hb = ((size_t)b * NH + wave) * VD;
        val[(hb + lm) * NN + row0 + m]        = f2bfu(acc0[g]);
        val[(hb + 16 + lm) * NN + row0 + m]   = f2bfu(acc1[g]);
    }
}

// =============================================================================
// Kernel 3: fused masked softmax + att@value + GELU.
// One wave per (b,h,16-row tile). A frag: 2x ds_load_b128 from bf16 prob tile.
// B frags: 4x global b128 from transposed value (L2-resident). f32 denominator
// accumulated during prob generation (matches reference softmax precision).
// =============================================================================
__global__ __launch_bounds__(256) void attn_kernel(
    const float* __restrict__ md, const unsigned short* __restrict__ val,
    const float* __restrict__ r, const float* __restrict__ thr,
    const float* __restrict__ dmin, float* __restrict__ out) {
    __shared__ unsigned short p_sh[NH][16][LROW];    // bf16 prob tile, per wave

    const int wave = threadIdx.x >> 5;
    const int lane = threadIdx.x & 31;
    const int gw   = blockIdx.x * 8 + wave;          // 0..2047
    const int mt   = gw & 127;
    const int h    = (gw >> 7) & 7;
    const int b    = gw >> 10;
    const int row0 = mt * 16;
    const int lm   = lane & 15;
    const int hi   = lane >> 4;
    const int base = hi << 3;

    const float scale = tanf(0.25f * 3.14159265358979323846f * (1.0f + sinf(r[h])));
    const float thr_m  = thr [b * NN + row0 + lm];
    const float dmin_m = dmin[b * NN + row0 + lm];

    const unsigned short* vrow0 = val + (((size_t)b * NH + h) * VD + lm) * NN;
    const unsigned short* vrow1 = val + (((size_t)b * NH + h) * VD + 16 + lm) * NN;

    v8f acc0 = {}, acc1 = {};
    float dsum[8] = {0.f, 0.f, 0.f, 0.f, 0.f, 0.f, 0.f, 0.f};

    for (int j0 = 0; j0 < NN; j0 += 32) {
        // prob tile: iteration tt covers rows 2*tt (lanes 0-15) and 2*tt+1 (16-31),
        // each lane handles cols 2*lm, 2*lm+1 -> 128B coalesced per row.
#pragma unroll
        for (int tt = 0; tt < 8; ++tt) {
            int rowi = 2 * tt + hi;
            float tv = __shfl(thr_m, rowi);
            float dv = __shfl(dmin_m, rowi);
            const float* rp = md + ((size_t)b * NN + row0 + rowi) * NN + j0 + 2 * lm;
            float2 d2 = *(const float2*)rp;
            if (j0 + 32 < NN) __builtin_prefetch(rp + 32, 0, 1);
            float p0 = (d2.x <= tv) ? __expf(-scale * (d2.x - dv)) : 0.0f;
            float p1 = (d2.y <= tv) ? __expf(-scale * (d2.y - dv)) : 0.0f;
            dsum[tt] += p0 + p1;                     // f32 softmax denominator
            unsigned int w = (unsigned int)f2bfu(p0) |
                             ((unsigned int)f2bfu(p1) << 16);
            *(unsigned int*)&p_sh[wave][rowi][2 * lm] = w;
        }
        lds_wave_fence();

        // A fragment: two contiguous 16B runs of row lm (K = base..base+7, +16)
        v16bf af = mk16(*(const uint4*)&p_sh[wave][lm][base],
                        *(const uint4*)&p_sh[wave][lm][16 + base]);
        // B fragments straight from transposed value in global (L2-resident)
        v16bf b0 = mk16(*(const uint4*)(vrow0 + j0 + base),
                        *(const uint4*)(vrow0 + j0 + 16 + base));
        v16bf b1 = mk16(*(const uint4*)(vrow1 + j0 + base),
                        *(const uint4*)(vrow1 + j0 + 16 + base));
        acc0 = wmma_bf16(af, b0, acc0);
        acc1 = wmma_bf16(af, b1, acc1);
        lds_wave_fence();                            // reads done before overwrite
    }

    // butterfly-reduce the 8 per-lane partials -> all 16 row denominators
    float den[16];
#pragma unroll
    for (int tt = 0; tt < 8; ++tt) {
        float s = dsum[tt];
        s += __shfl_xor(s, 1);
        s += __shfl_xor(s, 2);
        s += __shfl_xor(s, 4);
        s += __shfl_xor(s, 8);                       // total for row 2*tt+hi
        float o = __shfl_xor(s, 16);                 // total for row 2*tt+(1-hi)
        den[2 * tt]     = hi ? o : s;
        den[2 * tt + 1] = hi ? s : o;
    }

#pragma unroll
    for (int g = 0; g < 8; ++g) {
        int m = g + (hi << 3);
        float dm = hi ? den[g + 8] : den[g];
        float x0 = acc0[g] / dm;
        float x1 = acc1[g] / dm;
        size_t ob = ((size_t)b * NN + row0 + m) * HID + h * VD;
        out[ob + lm]      = gelu_exact(x0);
        out[ob + 16 + lm] = gelu_exact(x1);
    }
}

// =============================================================================
extern "C" void kernel_launch(void* const* d_in, const int* in_sizes, int n_in,
                              void* d_out, int out_size, void* d_ws, size_t ws_size,
                              hipStream_t stream) {
    (void)in_sizes; (void)n_in; (void)out_size; (void)ws_size;
    const float* md  = (const float*)d_in[0];   // [B,N,N]
    const float* inp = (const float*)d_in[1];   // [B,N,HID]
    const float* r   = (const float*)d_in[2];   // [1,H,1,1]
    const float* wgt = (const float*)d_in[3];   // [H,HID,V]
    float* out = (float*)d_out;                 // [B,N,HID] f32

    char* ws = (char*)d_ws;
    float*          thr  = (float*)ws;                                      // 16 KB
    float*          dmn  = (float*)(ws + (size_t)BB * NN * 4);              // 16 KB
    unsigned short* vval = (unsigned short*)(ws + (size_t)2 * BB * NN * 4); // 2 MB, [B,H,V,N] bf16

    row_median_min_kernel<<<BB * NN, 256, 0, stream>>>(md, thr, dmn);
    value_gemm_kernel<<<BB * (NN / 16), 256, 0, stream>>>(inp, wgt, vval);
    attn_kernel<<<(BB * NH * (NN / 16)) / 8, 256, 0, stream>>>(md, vval, r, thr, dmn, out);
}
